// ComprehensiveAngleInvariantAttention_52587579572725
// MI455X (gfx1250) — compile-verified, matrix operand order checked by reading, not run on hardware
//
#include <hip/hip_runtime.h>
#include <hip/hip_bf16.h>

typedef __bf16 bf16_t;
typedef __attribute__((ext_vector_type(16))) __bf16 v16bf;
typedef __attribute__((ext_vector_type(8)))  float  v8f;

#define B_    32
#define C_    256
#define N_    49
#define HEADS_ 8
#define NA_   16
#define KDIR  6400   // 256 * 25
#define NCHUNK (KDIR / 32)   // 200

// native f32 -> bf16 (gfx1250 has hardware bf16 converts; RTNE)
static __device__ __forceinline__ bf16_t f2bf(float f) { return (bf16_t)f; }

// fragment K mapping: slot jj (0..15), hi (0/1) -> kk in 0..31
// jj<8 : kk = hi*8 + jj ; jj>=8 : kk = hi*8 + jj + 8
static __device__ __forceinline__ int frag_kk(int jj, int hi) {
    return (jj & 7) + hi * 8 + ((jj >= 8) ? 16 : 0);
}

// ---------------------------------------------------------------------------
// Generic WMMA GEMM:  Y[b][m][n] = act( W[m,:] . X[b][:,n] + bias[m] ) + resid
// W: [M,K] row-major fp32 (shared across batch). X,Y,resid: [B][K|M][49] fp32.
// 4 waves per WG; wave w owns rows blockIdx.x*64 + w*16; cols 0..48 padded to
// 64 = four 16x16 WMMA column tiles. K must be a multiple of 32.
// A fragments: loaded straight from global (two 32B runs per lane, b128),
// row index clamped + select-zeroed (branch-free tail handling).
// B fragments: LDS in fragment-packed layout [slot(128)][16 bf16]; producer
// loads are unconditional (clamped column) + select-zero, one 32B LDS store.
// act: 0=none, 1=relu, 2=sigmoid
// ---------------------------------------------------------------------------
__global__ void __launch_bounds__(128)
gemm_bn_kernel(const float* __restrict__ Wm, const float* __restrict__ X,
               const float* __restrict__ bias, const float* __restrict__ resid,
               float* __restrict__ Y, int M, int K, int act)
{
    const int b    = blockIdx.y;
    const int tid  = threadIdx.x;
    const int lane = tid & 31;
    const int rowbase = blockIdx.x * 64 + (tid >> 5) * 16;
    const int row  = rowbase + (lane & 15);
    const int rowc = (row < M) ? row : (M - 1);      // clamped (always valid)
    const bool rowok = (row < M);
    const int hi   = lane >> 4;
    const int base = hi * 8;

    __shared__ __align__(32) bf16_t Bs[128 * 16];

    v8f acc0 = {0.f,0.f,0.f,0.f,0.f,0.f,0.f,0.f};
    v8f acc1 = acc0, acc2 = acc0, acc3 = acc0;

    const float* Xb = X + (size_t)b * K * N_;

    // this thread's B-producer slot
    const int sls = tid & 31;                        // lane-slot within tile
    const int sn  = (tid >> 5) * 16 + (sls & 15);    // logical column 0..63
    const int snc = (sn < N_) ? sn : (N_ - 1);       // clamped (always valid)
    const bool snok = (sn < N_);
    const int shi = sls >> 4;

    for (int k0 = 0; k0 < K; k0 += 32) {
        // ---- produce B tile (fragment-packed), branch-free ----
        {
            const float* xb0 = Xb + (size_t)k0 * N_ + snc;
            v16bf pk;
#pragma unroll
            for (int jj = 0; jj < 16; ++jj) {
                int kk = frag_kk(jj, shi);
                float v = xb0[kk * N_];              // unconditional load
                pk[jj] = f2bf(snok ? v : 0.f);       // select-zero pad columns
            }
            *(v16bf*)(&Bs[tid * 16]) = pk;
        }
        // ---- A fragment straight from global, branch-free ----
        v16bf af;
        {
            const float4* w4 = (const float4*)(Wm + (size_t)rowc * K + k0 + base);
            union { float4 v4[4]; float f[16]; } U;
            U.v4[0] = w4[0]; U.v4[1] = w4[1];        // kk = base .. base+7
            U.v4[2] = w4[4]; U.v4[3] = w4[5];        // kk = base+16 .. base+23
#pragma unroll
            for (int jj = 0; jj < 16; ++jj) af[jj] = f2bf(rowok ? U.f[jj] : 0.f);
        }
        __syncthreads();

#pragma unroll
        for (int j = 0; j < 4; ++j) {
            v16bf bfv = *(const v16bf*)(&Bs[(j * 32 + lane) * 16]);
            if (j == 0) acc0 = __builtin_amdgcn_wmma_f32_16x16x32_bf16(false, af, false, bfv, (short)0, acc0, false, false);
            if (j == 1) acc1 = __builtin_amdgcn_wmma_f32_16x16x32_bf16(false, af, false, bfv, (short)0, acc1, false, false);
            if (j == 2) acc2 = __builtin_amdgcn_wmma_f32_16x16x32_bf16(false, af, false, bfv, (short)0, acc2, false, false);
            if (j == 3) acc3 = __builtin_amdgcn_wmma_f32_16x16x32_bf16(false, af, false, bfv, (short)0, acc3, false, false);
        }
        __syncthreads();
    }

    float* Yb = Y + (size_t)b * M * N_;
    const float* Rb = resid ? resid + (size_t)b * M * N_ : nullptr;
#pragma unroll
    for (int j = 0; j < 4; ++j) {
        v8f a = (j == 0) ? acc0 : (j == 1) ? acc1 : (j == 2) ? acc2 : acc3;
        int n = j * 16 + (lane & 15);
#pragma unroll
        for (int r = 0; r < 8; ++r) {
            int rr = rowbase + r + (hi << 3);
            if (rr < M && n < N_) {
                float v = a[r];
                if (bias) v += bias[rr];
                if (act == 1) v = fmaxf(v, 0.f);
                else if (act == 2) v = 1.f / (1.f + expf(-v));
                if (Rb) v += Rb[(size_t)rr * N_ + n];
                Yb[(size_t)rr * N_ + n] = v;
            }
        }
    }
}

// ---------------------------------------------------------------------------
// Directional conv as GEMM with on-the-fly gate-folded weights. LDS-free.
// comb[b][o][n] = sum_k ( sum_a aw[b][a]*dirw[a][o][k] ) * patch[b][k][n]
// dirw: [16][256][6400] fp32; patches: fragment-packed bf16
// [B][chunk(200)][slot(128)][16]; aw: [B][16]
// ---------------------------------------------------------------------------
__global__ void __launch_bounds__(128)
dirconv_kernel(const float* __restrict__ dirw, const bf16_t* __restrict__ patches,
               const float* __restrict__ aw, float* __restrict__ Y)
{
    const int b    = blockIdx.y;
    const int tid  = threadIdx.x;
    const int lane = tid & 31;
    const int rowbase = blockIdx.x * 64 + (tid >> 5) * 16;   // < 256 always
    const int row  = rowbase + (lane & 15);
    const int hi   = lane >> 4;
    const int base = hi * 8;

    float awr[16];
#pragma unroll
    for (int a = 0; a < 16; ++a) awr[a] = aw[b * 16 + a];

    v8f acc0 = {0.f,0.f,0.f,0.f,0.f,0.f,0.f,0.f};
    v8f acc1 = acc0, acc2 = acc0, acc3 = acc0;

    const bf16_t* Pb = patches + (size_t)b * NCHUNK * 128 * 16;

    for (int chunk = 0; chunk < NCHUNK; ++chunk) {
        const int k0 = chunk * 32;
        // ---- fold 16 angle weights into this lane's A fragment (registers) ----
        float fr[16];
#pragma unroll
        for (int t = 0; t < 16; ++t) fr[t] = 0.f;
        for (int a = 0; a < 16; ++a) {
            const float s = awr[a];
            const float4* w4 = (const float4*)(dirw + ((size_t)a * C_ + row) * KDIR + k0 + base);
            union { float4 v4[4]; float f[16]; } U;
            U.v4[0] = w4[0]; U.v4[1] = w4[1];   // kk = base .. base+7
            U.v4[2] = w4[4]; U.v4[3] = w4[5];   // kk = base+16 .. base+23
#pragma unroll
            for (int t = 0; t < 16; ++t) fr[t] += s * U.f[t];
        }
        v16bf af;
#pragma unroll
        for (int jj = 0; jj < 16; ++jj) af[jj] = f2bf(fr[jj]);

        // ---- B fragments: coalesced 32B global loads (packed layout) ----
        const bf16_t* Pc = Pb + (size_t)chunk * (128 * 16);
#pragma unroll
        for (int j = 0; j < 4; ++j) {
            v16bf bfv = *(const v16bf*)(Pc + (size_t)(j * 32 + lane) * 16);
            if (j == 0) acc0 = __builtin_amdgcn_wmma_f32_16x16x32_bf16(false, af, false, bfv, (short)0, acc0, false, false);
            if (j == 1) acc1 = __builtin_amdgcn_wmma_f32_16x16x32_bf16(false, af, false, bfv, (short)0, acc1, false, false);
            if (j == 2) acc2 = __builtin_amdgcn_wmma_f32_16x16x32_bf16(false, af, false, bfv, (short)0, acc2, false, false);
            if (j == 3) acc3 = __builtin_amdgcn_wmma_f32_16x16x32_bf16(false, af, false, bfv, (short)0, acc3, false, false);
        }
    }

    float* Yb = Y + (size_t)b * C_ * N_;
#pragma unroll
    for (int j = 0; j < 4; ++j) {
        v8f a = (j == 0) ? acc0 : (j == 1) ? acc1 : (j == 2) ? acc2 : acc3;
        int n = j * 16 + (lane & 15);
#pragma unroll
        for (int r = 0; r < 8; ++r) {
            int rr = rowbase + r + (hi << 3);
            if (n < N_) Yb[(size_t)rr * N_ + n] = a[r];
        }
    }
}

// ---------------------------------------------------------------------------
// im2col patches in fragment-packed order:
// patches[b][chunk][slot][jj] = normed[b][i][y+dy, x+dx] or 0, where
// slot -> (j = slot>>5, ls = slot&31, n = j*16+(ls&15), hi = ls>>4),
// k = chunk*32 + frag_kk(jj, hi), i = k/25, p = k%25.
// ---------------------------------------------------------------------------
__global__ void patches_kernel(const float* __restrict__ normed, bf16_t* __restrict__ patches)
{
    int tid = blockIdx.x * blockDim.x + threadIdx.x;
    const int total = B_ * NCHUNK * 128 * 16;
    if (tid >= total) return;
    int jj    = tid & 15;
    int slot  = (tid >> 4) & 127;
    int chunk = (tid >> 11) % NCHUNK;
    int b     = tid / (NCHUNK * 2048);
    int j  = slot >> 5;
    int ls = slot & 31;
    int n  = j * 16 + (ls & 15);
    int hi = ls >> 4;
    int k  = chunk * 32 + frag_kk(jj, hi);
    float v = 0.f;
    if (n < N_) {
        int i = k / 25, p = k % 25;
        int y = n / 7 + p / 5 - 2;
        int x = n % 7 + p % 5 - 2;
        if (y >= 0 && y < 7 && x >= 0 && x < 7)
            v = normed[((size_t)b * C_ + i) * N_ + y * 7 + x];
    }
    patches[tid] = f2bf(v);
}

// ---------------------------------------------------------------------------
// GroupNorm (32 groups of 8 ch x 49).  If mul != null, value = x * (1 + mul)
// ---------------------------------------------------------------------------
__global__ void __launch_bounds__(128)
gn_kernel(const float* __restrict__ X, const float* __restrict__ mul,
          const float* __restrict__ g, const float* __restrict__ be, float* __restrict__ Y)
{
    const int b   = blockIdx.x >> 5;
    const int grp = blockIdx.x & 31;
    const int tid = threadIdx.x;
    const size_t off = ((size_t)b * C_ + grp * 8) * N_;
    const float* xb = X + off;
    const float* mb = mul ? mul + off : nullptr;
    __shared__ float s1[128], s2[128];
    __shared__ float mv[2];
    float s = 0.f, q = 0.f;
    for (int e = tid; e < 392; e += 128) {
        float v = xb[e]; if (mb) v *= (1.f + mb[e]);
        s += v; q += v * v;
    }
    s1[tid] = s; s2[tid] = q; __syncthreads();
    for (int st = 64; st > 0; st >>= 1) {
        if (tid < st) { s1[tid] += s1[tid + st]; s2[tid] += s2[tid + st]; }
        __syncthreads();
    }
    if (tid == 0) {
        float m = s1[0] / 392.f;
        float var = s2[0] / 392.f - m * m;
        mv[0] = m; mv[1] = rsqrtf(var + 1e-5f);
    }
    __syncthreads();
    float m = mv[0], inv = mv[1];
    for (int e = tid; e < 392; e += 128) {
        int c = grp * 8 + e / N_;
        float v = xb[e]; if (mb) v *= (1.f + mb[e]);
        Y[off + e] = (v - m) * inv * g[c] + be[c];
    }
}

// LayerNorm over channels per token: in/out layout [B][C][49]
__global__ void __launch_bounds__(128)
ln_kernel(const float* __restrict__ X, const float* __restrict__ g,
          const float* __restrict__ be, float* __restrict__ Y)
{
    const int b = blockIdx.x / N_, n = blockIdx.x % N_;
    const int tid = threadIdx.x;
    float v0 = X[((size_t)b * C_ + tid) * N_ + n];
    float v1 = X[((size_t)b * C_ + tid + 128) * N_ + n];
    __shared__ float s1[128], s2[128];
    __shared__ float mv[2];
    s1[tid] = v0 + v1; s2[tid] = v0 * v0 + v1 * v1; __syncthreads();
    for (int st = 64; st > 0; st >>= 1) {
        if (tid < st) { s1[tid] += s1[tid + st]; s2[tid] += s2[tid + st]; }
        __syncthreads();
    }
    if (tid == 0) {
        float m = s1[0] / 256.f;
        float var = s2[0] / 256.f - m * m;
        mv[0] = m; mv[1] = rsqrtf(var + 1e-5f);
    }
    __syncthreads();
    float m = mv[0], inv = mv[1];
    Y[((size_t)b * C_ + tid) * N_ + n]       = (v0 - m) * inv * g[tid] + be[tid];
    Y[((size_t)b * C_ + tid + 128) * N_ + n] = (v1 - m) * inv * g[tid + 128] + be[tid + 128];
}

// ---------------------------------------------------------------------------
// Softmax attention over 49 tokens, per (batch, head).
// qkv: [B][3*heads*d][49]; out: [B][heads*d][49]; bias (opt): [heads][49][49]
// ---------------------------------------------------------------------------
__global__ void __launch_bounds__(64)
attn_kernel(const float* __restrict__ qkv, const float* __restrict__ bias,
            float* __restrict__ out, int d, int heads, float scale)
{
    const int b = blockIdx.x / heads, h = blockIdx.x % heads;
    const int tid = threadIdx.x;
    const int C3 = 3 * heads * d;
    const float* base = qkv + (size_t)b * C3 * N_;
    __shared__ float sq[64 * N_], sk[64 * N_], sv[64 * N_];
    __shared__ float sc[N_ * N_];
    const int qo = h * d, ko = heads * d + h * d, vo = 2 * heads * d + h * d;
    for (int idx = tid; idx < d * N_; idx += 64) {
        sq[idx] = base[(size_t)qo * N_ + idx];
        sk[idx] = base[(size_t)ko * N_ + idx];
        sv[idx] = base[(size_t)vo * N_ + idx];
    }
    __syncthreads();
    if (tid < N_) {
        const int i = tid;
        float mx = -1e30f;
        for (int j = 0; j < N_; ++j) {
            float s = 0.f;
            for (int dd = 0; dd < d; ++dd) s += sq[dd * N_ + i] * sk[dd * N_ + j];
            s *= scale;
            if (bias) s += bias[(size_t)h * N_ * N_ + i * N_ + j];
            sc[i * N_ + j] = s;
            mx = fmaxf(mx, s);
        }
        float sum = 0.f;
        for (int j = 0; j < N_; ++j) { float e = expf(sc[i * N_ + j] - mx); sc[i * N_ + j] = e; sum += e; }
        float inv = 1.f / sum;
        for (int j = 0; j < N_; ++j) sc[i * N_ + j] *= inv;
    }
    __syncthreads();
    float* ob = out + ((size_t)b * heads + h) * d * N_;
    for (int idx = tid; idx < d * N_; idx += 64) {
        int dd = idx / N_, i = idx % N_;
        float s = 0.f;
        for (int j = 0; j < N_; ++j) s += sc[i * N_ + j] * sv[dd * N_ + j];
        ob[idx] = s;
    }
}

// Rel-pos bias table expansion: biasbuf[h][i][j] = relpos[dy][dx][h]
__global__ void relpos_kernel(const float* __restrict__ relpos, float* __restrict__ biasbuf)
{
    int tid = blockIdx.x * blockDim.x + threadIdx.x;
    if (tid >= HEADS_ * N_ * N_) return;
    int h = tid / (N_ * N_);
    int i = (tid / N_) % N_;
    int j = tid % N_;
    int dy = j / 7 - i / 7 + 6;
    int dx = j % 7 - i % 7 + 6;
    biasbuf[tid] = relpos[(dy * 13 + dx) * 8 + h];
}

// GAP -> MLP(128,relu) -> 16-way softmax gate:  aw[b][16]
__global__ void __launch_bounds__(128)
aw_kernel(const float* __restrict__ normed, const float* __restrict__ w1, const float* __restrict__ b1,
          const float* __restrict__ w2, const float* __restrict__ b2, float* __restrict__ aw)
{
    const int b = blockIdx.x, tid = threadIdx.x;
    __shared__ float gap[C_], h1[128], lg[16], red[2];
    for (int c = tid; c < C_; c += 128) {
        float s = 0.f;
        const float* p = normed + ((size_t)b * C_ + c) * N_;
        for (int n = 0; n < N_; ++n) s += p[n];
        gap[c] = s * (1.f / N_);
    }
    __syncthreads();
    {
        float s = b1[tid];
        const float* w = w1 + (size_t)tid * C_;
        for (int c = 0; c < C_; ++c) s += gap[c] * w[c];
        h1[tid] = fmaxf(s, 0.f);
    }
    __syncthreads();
    if (tid < 16) {
        float s = b2[tid];
        const float* w = w2 + (size_t)tid * 128;
        for (int j = 0; j < 128; ++j) s += h1[j] * w[j];
        lg[tid] = s;
    }
    __syncthreads();
    if (tid == 0) {
        float mx = lg[0];
        for (int a = 1; a < 16; ++a) mx = fmaxf(mx, lg[a]);
        float sum = 0.f;
        for (int a = 0; a < 16; ++a) sum += expf(lg[a] - mx);
        red[0] = mx; red[1] = 1.f / sum;
    }
    __syncthreads();
    if (tid < 16) aw[b * 16 + tid] = expf(lg[tid] - red[0]) * red[1];
}

// 4-way gate softmax + weighted branch fusion
__global__ void fuse_kernel(const float* __restrict__ sp, const float* __restrict__ rot,
                            const float* __restrict__ ang, const float* __restrict__ ada,
                            const float* __restrict__ glog, float* __restrict__ fused)
{
    int tid = blockIdx.x * blockDim.x + threadIdx.x;
    if (tid >= B_ * C_ * N_) return;
    int n = tid % N_;
    int b = tid / (C_ * N_);
    const float* g = glog + (size_t)b * 4 * N_;
    float g0 = g[0 * N_ + n], g1 = g[1 * N_ + n], g2 = g[2 * N_ + n], g3 = g[3 * N_ + n];
    float m = fmaxf(fmaxf(g0, g1), fmaxf(g2, g3));
    float e0 = expf(g0 - m), e1 = expf(g1 - m), e2 = expf(g2 - m), e3 = expf(g3 - m);
    float inv = 1.f / (e0 + e1 + e2 + e3);
    fused[tid] = (sp[tid] * e0 + rot[tid] * e1 + ang[tid] * e2 + ada[tid] * e3) * inv;
}

// ---------------------------------------------------------------------------
extern "C" void kernel_launch(void* const* d_in, const int* in_sizes, int n_in,
                              void* d_out, int out_size, void* d_ws, size_t ws_size,
                              hipStream_t stream)
{
    const float* x        = (const float*)d_in[0];
    const float* gn_g     = (const float*)d_in[1];
    const float* gn_b     = (const float*)d_in[2];
    const float* sa_w1    = (const float*)d_in[3];
    const float* sa_b1    = (const float*)d_in[4];
    const float* sa_w2    = (const float*)d_in[5];
    const float* sa_b2    = (const float*)d_in[6];
    const float* sa_gn_g  = (const float*)d_in[7];
    const float* sa_gn_b  = (const float*)d_in[8];
    const float* ra_gn_g  = (const float*)d_in[9];
    const float* ra_gn_b  = (const float*)d_in[10];
    const float* ra_qkv_w = (const float*)d_in[11];
    const float* ra_proj_w= (const float*)d_in[12];
    const float* ra_proj_b= (const float*)d_in[13];
    const float* aa_ln_g  = (const float*)d_in[14];
    const float* aa_ln_b  = (const float*)d_in[15];
    const float* aa_qkv_w = (const float*)d_in[16];
    const float* aa_relpos= (const float*)d_in[17];
    const float* aa_out_w = (const float*)d_in[18];
    const float* aa_out_b = (const float*)d_in[19];
    const float* ad_dir_w = (const float*)d_in[20];
    const float* ad_ap_w1 = (const float*)d_in[21];
    const float* ad_ap_b1 = (const float*)d_in[22];
    const float* ad_ap_w2 = (const float*)d_in[23];
    const float* ad_ap_b2 = (const float*)d_in[24];
    const float* ad_proj_w= (const float*)d_in[25];
    const float* ad_proj_b= (const float*)d_in[26];
    const float* gate_w   = (const float*)d_in[27];
    const float* gate_b   = (const float*)d_in[28];
    const float* proj_w   = (const float*)d_in[29];
    const float* proj_b   = (const float*)d_in[30];

    float* ws = (float*)d_ws;
    size_t o = 0;
    const size_t BCN = (size_t)B_ * C_ * N_;
    float* normed   = ws + o; o += BCN;
    float* h1sa     = ws + o; o += (size_t)B_ * 32 * N_;
    float* wts      = ws + o; o += BCN;
    float* spatial  = ws + o; o += BCN;
    float* xr       = ws + o; o += BCN;
    float* qkv_ra   = ws + o; o += (size_t)B_ * 768 * N_;
    float* o_ra     = ws + o; o += BCN;
    float* rotation = ws + o; o += BCN;
    float* tns      = ws + o; o += BCN;
    float* qkv_aa   = ws + o; o += (size_t)B_ * 1536 * N_;
    float* biasbuf  = ws + o; o += (size_t)HEADS_ * N_ * N_;
    float* o_aa     = ws + o; o += (size_t)B_ * 512 * N_;
    float* angle    = ws + o; o += BCN;
    float* awb      = ws + o; o += (size_t)B_ * 16;
    float* comb     = ws + o; o += BCN;
    float* adaptive = ws + o; o += BCN;
    float* glog     = ws + o; o += (size_t)B_ * 4 * N_;
    float* fused    = ws + o; o += BCN;
    o = (o + 7) & ~(size_t)7;                   // 32B align for v16bf loads
    bf16_t* patches = (bf16_t*)(ws + o);        // [B][200][128][16] bf16 (26.2 MB)

    float* outp = (float*)d_out;

    // --- base GroupNorm ---
    gn_kernel<<<B_ * 32, 128, 0, stream>>>(x, nullptr, gn_g, gn_b, normed);

    // --- spatial (SE) branch ---
    gemm_bn_kernel<<<dim3(1, B_), 128, 0, stream>>>(sa_w1, normed, sa_b1, nullptr, h1sa, 32, 256, 1);
    gemm_bn_kernel<<<dim3(4, B_), 128, 0, stream>>>(sa_w2, h1sa, sa_b2, nullptr, wts, 256, 32, 2);
    gn_kernel<<<B_ * 32, 128, 0, stream>>>(normed, wts, sa_gn_g, sa_gn_b, spatial);

    // --- rotation branch (rot90 passes are permutation-equivariant => one pass) ---
    gn_kernel<<<B_ * 32, 128, 0, stream>>>(normed, nullptr, ra_gn_g, ra_gn_b, xr);
    gemm_bn_kernel<<<dim3(12, B_), 128, 0, stream>>>(ra_qkv_w, xr, nullptr, nullptr, qkv_ra, 768, 256, 0);
    attn_kernel<<<B_ * HEADS_, 64, 0, stream>>>(qkv_ra, nullptr, o_ra, 32, HEADS_, 0.1767766953f);
    gemm_bn_kernel<<<dim3(4, B_), 128, 0, stream>>>(ra_proj_w, o_ra, ra_proj_b, normed, rotation, 256, 256, 0);

    // --- angle-aware MHSA branch ---
    ln_kernel<<<B_ * N_, 128, 0, stream>>>(normed, aa_ln_g, aa_ln_b, tns);
    gemm_bn_kernel<<<dim3(24, B_), 128, 0, stream>>>(aa_qkv_w, tns, nullptr, nullptr, qkv_aa, 1536, 256, 0);
    relpos_kernel<<<(HEADS_ * N_ * N_ + 255) / 256, 256, 0, stream>>>(aa_relpos, biasbuf);
    attn_kernel<<<B_ * HEADS_, 64, 0, stream>>>(qkv_aa, biasbuf, o_aa, 64, HEADS_, 0.125f);
    gemm_bn_kernel<<<dim3(4, B_), 128, 0, stream>>>(aa_out_w, o_aa, aa_out_b, normed, angle, 256, 512, 0);

    // --- angle-adaptive branch (gate folded into conv weights) ---
    aw_kernel<<<B_, 128, 0, stream>>>(normed, ad_ap_w1, ad_ap_b1, ad_ap_w2, ad_ap_b2, awb);
    patches_kernel<<<(B_ * NCHUNK * 2048 + 255) / 256, 256, 0, stream>>>(normed, patches);
    dirconv_kernel<<<dim3(4, B_), 128, 0, stream>>>(ad_dir_w, patches, awb, comb);
    gemm_bn_kernel<<<dim3(4, B_), 128, 0, stream>>>(ad_proj_w, comb, ad_proj_b, normed, adaptive, 256, 256, 0);

    // --- gated fusion + final projection ---
    gemm_bn_kernel<<<dim3(1, B_), 128, 0, stream>>>(gate_w, x, gate_b, nullptr, glog, 4, 256, 0);
    fuse_kernel<<<((int)BCN + 255) / 256, 256, 0, stream>>>(spatial, rotation, angle, adaptive, glog, fused);
    gemm_bn_kernel<<<dim3(4, B_), 128, 0, stream>>>(proj_w, fused, proj_b, x, outp, 256, 256, 0);
}
